// Ithemal_58969900974490
// MI455X (gfx1250) — compile-verified, matrix operand order checked by reading, not run on hardware
//
#include <hip/hip_runtime.h>
#include <hip/hip_bf16.h>

// CDNA5 / gfx1250 wave32 WMMA implementation of Ithemal two-level LSTM.
// GEMM core: v_wmma_f32_16x16x32_bf16, fp32 accumulate, bf16 operands.
// Weights streamed from L2 each step (2MB bf16 total, L2-resident);
// an opaque per-iteration offset stops LICM from hoisting+spilling them.

typedef __attribute__((ext_vector_type(16))) __bf16 v16bf;
typedef __attribute__((ext_vector_type(8)))  float  v8f;

#define HID    256
#define KTILES 8      // 256 / 32 (K per WMMA = 32)
#define NTILES 64     // 1024 gate cols / 16
#define FRAG_USH 512  // 32 lanes * 16 bf16 per fragment

__device__ __forceinline__ unsigned short f2bf(float x) {
  unsigned u = __float_as_uint(x);
  u += 0x7FFFu + ((u >> 16) & 1u);   // round-to-nearest-even
  return (unsigned short)(u >> 16);
}
__device__ __forceinline__ float sigm(float x) { return 1.0f / (1.0f + __expf(-x)); }

// ---------------------------------------------------------------------------
// Weight swizzle: W[4H=1024][K=256] fp32 -> bf16 B-fragments
//   frag[kt][nt][lane][j]:  column n = nt*16 + (lane&15)
//   k = kt*32 + ((j<8)? j : j+8) + (lane>=16 ? 8 : 0)
// ---------------------------------------------------------------------------
__global__ __launch_bounds__(256) void convert_weights(
    const float* __restrict__ w0, const float* __restrict__ w1,
    const float* __restrict__ w2, const float* __restrict__ w3,
    unsigned short* __restrict__ outw)
{
  int idx = blockIdx.x * 256 + threadIdx.x;   // exactly 4 * 262144 threads
  int mat = idx >> 18;
  int r   = idx & 0x3FFFF;
  int j    = r & 15;
  int lane = (r >> 4) & 31;
  int nt   = (r >> 9) & 63;
  int kt   = (r >> 15) & 7;
  int n = nt * 16 + (lane & 15);
  int k = kt * 32 + ((j < 8) ? j : j + 8) + ((lane >= 16) ? 8 : 0);
  const float* W = (mat == 0) ? w0 : (mat == 1) ? w1 : (mat == 2) ? w2 : w3;
  outw[idx] = f2bf(W[(size_t)n * 256 + k]);
}

__global__ __launch_bounds__(256) void zero_ws(float* __restrict__ p, int n) {
  int i = blockIdx.x * 256 + threadIdx.x;
  if (i < n) p[i] = 0.0f;
}

// ---------------------------------------------------------------------------
// One LSTM level. 256 threads = 8 waves; workgroup owns RT row-tiles of 16
// sequences. Wave w owns hidden columns [32w, 32w+32): per gate (i,f,g,o)
// that is 2 output tiles of 16x16 per row-tile -> 8*RT accumulators,
// 128*RT WMMAs/step; each B fragment load feeds RT WMMAs.
// ---------------------------------------------------------------------------
template <int RT>
__global__ __launch_bounds__(256) void lstm_level(
    int level,
    const int*   __restrict__ blocks,  // [B*I,16] tokens       (level 0)
    const float* __restrict__ emb,     // [V,256]               (level 0)
    const float* __restrict__ xsrc,    // [B,64,256] instr_repr (level 1)
    const int*   __restrict__ lens,    // per-sequence lengths
    const unsigned short* __restrict__ WihB,  // swizzled bf16 [8][64][32][16]
    const unsigned short* __restrict__ WhhB,
    const float* __restrict__ bias,    // [1024] torch gate order i,f,g,o
    float*       __restrict__ out,     // [nseq,256] h at len-1
    int Tsteps)
{
  __shared__ __align__(32) unsigned short xA[RT * KTILES * FRAG_USH];
  __shared__ __align__(32) unsigned short hA[RT * KTILES * FRAG_USH];
  __shared__ int lensS[RT * 16];

  const int tid    = threadIdx.x;
  const int wg     = blockIdx.x;
  const int s0     = wg * 16 * RT;
  const int w      = tid >> 5;      // wave id 0..7
  const int lane   = tid & 31;
  const int lane15 = lane & 15;
  const int mhi    = (lane >= 16) ? 8 : 0;

  if (tid < RT * 16) lensS[tid] = lens[s0 + tid];
  #pragma unroll
  for (int q = 0; q < RT * 16; ++q) hA[tid * (RT * 16) + q] = 0;   // h0 = 0

  // per-lane bias, hoisted out of the time loop
  float bI[2], bF[2], bG[2], bO[2];
  #pragma unroll
  for (int s2 = 0; s2 < 2; ++s2) {
    int n = w * 32 + s2 * 16 + lane15;
    bI[s2] = bias[0 * HID + n];
    bF[s2] = bias[1 * HID + n];
    bG[s2] = bias[2 * HID + n];
    bO[s2] = bias[3 * HID + n];
  }

  float cc[RT][2][8];                                   // c0 = 0
  #pragma unroll
  for (int rt = 0; rt < RT; ++rt)
    #pragma unroll
    for (int s2 = 0; s2 < 2; ++s2)
      #pragma unroll
      for (int v = 0; v < 8; ++v) cc[rt][s2][v] = 0.0f;

  for (int t = 0; t < Tsteps; ++t) {
    // ---- gather x_t (RT*16 rows x 256 f32) -> swizzled bf16 A fragments ----
    {
      const int r  = tid >> 4;
      const int kb = tid & 15;
      #pragma unroll
      for (int rt = 0; rt < RT; ++rt) {
        const int row = rt * 16 + r;
        const float* src;
        if (level == 0) {
          int tok = blocks[(size_t)(s0 + row) * 16 + t];
          src = emb + (size_t)tok * HID;
        } else {
          src = xsrc + ((size_t)(s0 + row) * 64 + t) * HID;
        }
        #pragma unroll
        for (int q = 0; q < 16; ++q) {
          int k = kb + q * 16;                 // coalesced across lanes
          float val = src[k];
          int kt = k >> 5, ko = k & 31;
          int l = r + ((ko & 8) ? 16 : 0);
          int j = ((ko & 16) ? 8 : 0) + (ko & 7);
          xA[rt * (KTILES * FRAG_USH) + (kt * 32 + l) * 16 + j] = f2bf(val);
        }
      }
    }
    __syncthreads();

    // Opaque zero: makes weight addresses loop-variant so LICM cannot hoist
    // the 128 B-fragment loads out of the time loop (round-1 scratch spills).
    int koffs = 0;
    asm volatile("" : "+v"(koffs));

    // ---- gates = x_t @ Wih^T + h_{t-1} @ Whh^T  (K = 256 + 256) ----
    v8f acc[8][RT];
    #pragma unroll
    for (int i = 0; i < 8; ++i)
      #pragma unroll
      for (int rt = 0; rt < RT; ++rt)
        acc[i][rt] = (v8f){0.f,0.f,0.f,0.f,0.f,0.f,0.f,0.f};

    for (int kt = 0; kt < KTILES; ++kt) {
      v16bf a[RT];
      #pragma unroll
      for (int rt = 0; rt < RT; ++rt)
        a[rt] = *(const v16bf*)&xA[rt * (KTILES * FRAG_USH) + (kt * 32 + lane) * 16];
      #pragma unroll
      for (int gs0 = 0; gs0 < 8; gs0 += 4) {
        v16bf bm[4];
        #pragma unroll
        for (int u = 0; u < 4; ++u) {
          int gs = gs0 + u;
          int nt = (gs >> 1) * 16 + w * 2 + (gs & 1);
          bm[u] = *(const v16bf*)&WihB[((kt * NTILES + nt) * 32 + lane) * 16 + koffs];
        }
        #pragma unroll
        for (int u = 0; u < 4; ++u)
          #pragma unroll
          for (int rt = 0; rt < RT; ++rt)
            acc[gs0 + u][rt] = __builtin_amdgcn_wmma_f32_16x16x32_bf16(
                false, a[rt], false, bm[u], (short)0, acc[gs0 + u][rt], false, false);
      }
    }
    for (int kt = 0; kt < KTILES; ++kt) {
      v16bf a[RT];
      #pragma unroll
      for (int rt = 0; rt < RT; ++rt)
        a[rt] = *(const v16bf*)&hA[rt * (KTILES * FRAG_USH) + (kt * 32 + lane) * 16];
      #pragma unroll
      for (int gs0 = 0; gs0 < 8; gs0 += 4) {
        v16bf bm[4];
        #pragma unroll
        for (int u = 0; u < 4; ++u) {
          int gs = gs0 + u;
          int nt = (gs >> 1) * 16 + w * 2 + (gs & 1);
          bm[u] = *(const v16bf*)&WhhB[((kt * NTILES + nt) * 32 + lane) * 16 + koffs];
        }
        #pragma unroll
        for (int u = 0; u < 4; ++u)
          #pragma unroll
          for (int rt = 0; rt < RT; ++rt)
            acc[gs0 + u][rt] = __builtin_amdgcn_wmma_f32_16x16x32_bf16(
                false, a[rt], false, bm[u], (short)0, acc[gs0 + u][rt], false, false);
      }
    }

    // ---- LSTM cell (elementwise, fp32). C-layout: lane -> N, vgpr -> M ----
    float hval[RT][2][8];
    #pragma unroll
    for (int rt = 0; rt < RT; ++rt)
      #pragma unroll
      for (int s2 = 0; s2 < 2; ++s2)
        #pragma unroll
        for (int v = 0; v < 8; ++v) {
          float iv = sigm(acc[0 + s2][rt][v] + bI[s2]);
          float fv = sigm(acc[2 + s2][rt][v] + bF[s2]);
          float gv = tanhf(acc[4 + s2][rt][v] + bG[s2]);
          float ov = sigm(acc[6 + s2][rt][v] + bO[s2]);
          float cv = fv * cc[rt][s2][v] + iv * gv;
          cc[rt][s2][v] = cv;
          hval[rt][s2][v] = ov * tanhf(cv);
        }

    __syncthreads();   // all waves done reading xA(t)/hA(t-1)

    // ---- write h_t back as bf16 fragments (wave w owns k-tile w), snapshot ----
    #pragma unroll
    for (int rt = 0; rt < RT; ++rt)
      #pragma unroll
      for (int s2 = 0; s2 < 2; ++s2)
        #pragma unroll
        for (int v = 0; v < 8; ++v) {
          int m    = v + mhi;                       // row within 16-seq tile
          int ncol = w * 32 + s2 * 16 + lane15;     // hidden column
          int ko   = ncol & 31;
          int l = m + ((ko & 8) ? 16 : 0);
          int j = ((ko & 16) ? 8 : 0) + (ko & 7);
          hA[rt * (KTILES * FRAG_USH) + (w * 32 + l) * 16 + j] = f2bf(hval[rt][s2][v]);
          if (t == lensS[rt * 16 + m] - 1)
            out[(size_t)(s0 + rt * 16 + m) * HID + ncol] = hval[rt][s2][v];
        }
    // next iteration's gather-barrier orders hA writes vs. next compute reads
  }
}

// ---------------------------------------------------------------------------
// out[b] = dot(block_repr[b], linW) + linb
// ---------------------------------------------------------------------------
__global__ __launch_bounds__(256) void linear_kernel(
    const float* __restrict__ br, const float* __restrict__ linW,
    const float* __restrict__ linb, float* __restrict__ out)
{
  __shared__ float red[256];
  int b = blockIdx.x, tid = threadIdx.x;
  red[tid] = br[(size_t)b * HID + tid] * linW[tid];
  __syncthreads();
  for (int s = 128; s > 0; s >>= 1) {
    if (tid < s) red[tid] += red[tid + s];
    __syncthreads();
  }
  if (tid == 0) out[b] = red[0] + linb[0];
}

// ---------------------------------------------------------------------------
extern "C" void kernel_launch(void* const* d_in, const int* in_sizes, int n_in,
                              void* d_out, int out_size, void* d_ws, size_t ws_size,
                              hipStream_t stream) {
  (void)in_sizes; (void)n_in; (void)out_size; (void)ws_size;
  const int*   blocks       = (const int*)  d_in[0];   // [256,64,16]
  const int*   instr_counts = (const int*)  d_in[1];   // [256]
  const int*   token_counts = (const int*)  d_in[2];   // [256,64]
  const float* emb          = (const float*)d_in[3];   // [4096,256]
  const float* Wih_tok      = (const float*)d_in[4];
  const float* Whh_tok      = (const float*)d_in[5];
  const float* b_tok        = (const float*)d_in[6];
  const float* Wih_ins      = (const float*)d_in[7];
  const float* Whh_ins      = (const float*)d_in[8];
  const float* b_ins        = (const float*)d_in[9];
  const float* linW         = (const float*)d_in[10];
  const float* linb         = (const float*)d_in[11];
  float* out = (float*)d_out;                          // [256,1] f32

  // Workspace map:
  //   [0, 2MB)        4 swizzled bf16 weight mats (262144 ushorts each)
  //   [2MB, 18MB)     instr_repr fp32 [16384, 256]
  //   [18MB, 18.25MB) block_repr fp32 [256, 256]
  unsigned short* wbf = (unsigned short*)d_ws;
  float* instr_repr = (float*)((char*)d_ws + (size_t)2 * 1024 * 1024);
  float* block_repr = instr_repr + (size_t)16384 * 256;

  convert_weights<<<4096, 256, 0, stream>>>(Wih_tok, Whh_tok, Wih_ins, Whh_ins, wbf);

  int nz = 16384 * 256 + 256 * 256;
  zero_ws<<<(nz + 255) / 256, 256, 0, stream>>>(instr_repr, nz);

  // level 0: token LSTM over 16384 sequences (32 rows / WG), T = 16
  lstm_level<2><<<512, 256, 0, stream>>>(0, blocks, emb, nullptr, token_counts,
                                         wbf + 0, wbf + 262144, b_tok,
                                         instr_repr, 16);
  // level 1: instruction LSTM over 256 sequences (16 rows / WG), T = 64
  lstm_level<1><<<16, 256, 0, stream>>>(1, nullptr, nullptr, instr_repr, instr_counts,
                                        wbf + 524288, wbf + 786432, b_ins,
                                        block_repr, 64);

  linear_kernel<<<256, 256, 0, stream>>>(block_repr, linW, linb, out);
}
